// BGRU2dLayer_90177133346849
// MI455X (gfx1250) — matrix operand order repaired and checked by reading, not compiled, not run on hardware
//
#include <hip/hip_runtime.h>
#include <hip/hip_bf16.h>

typedef __attribute__((ext_vector_type(16))) __bf16 v16bf;
typedef __attribute__((ext_vector_type(8)))  __bf16 v8bf;
typedef __attribute__((ext_vector_type(8)))  float  v8f;

// Problem constants (B=16, T0=T1=64, E=128, H=128)
#define NB   16
#define NT   64
#define NH   128
#define N4H  512
#define N2H  256

__device__ __forceinline__ float sigmoidf_(float x) { return 1.0f / (1.0f + __expf(-x)); }

union BFrag { v16bf v; struct { v8bf lo, hi; } p; };

// ---- WMMA fragment loaders (per CDNA5 ISA layout tables) -------------------
// 16-bit A matrix 16x32 (MxK):
//  lanes 0-15 : M=lane,    elems 0..7 -> K=k0+0..7,  elems 8..15 -> K=k0+16..23
//  lanes16-31 : M=lane-16, elems 0..7 -> K=k0+8..15, elems 8..15 -> K=k0+24..31
// Both 8-element groups are contiguous -> two 16B ds_load_b128.
__device__ __forceinline__ v16bf load_a_frag(const __bf16* A, int stride, int k0, int lane) {
    const int m    = lane & 15;
    const int half = lane >> 4;
    const __bf16* row = A + m * stride + k0 + half * 8;
    BFrag u;
    u.p.lo = *(const v8bf*)(row);
    u.p.hi = *(const v8bf*)(row + 16);
    return u.v;
}

// 16-bit B matrix 32x16 (KxN): lane holds column n = n0+(lane&15);
//  lanes 0-15 hold K=0..7,16..23, lanes 16-31 hold K=8..15,24..31 of the chunk.
// Source is staged N-MAJOR in LDS (Bn[n][k], strideElems per n-row), so the
// each 8-element K group per lane is contiguous -> two 16B ds_load_b128.
__device__ __forceinline__ v16bf load_b_frag_nmaj(const __bf16* Bn, int strideElems,
                                                  int n0, int lane) {
    const __bf16* p = Bn + (size_t)(n0 + (lane & 15)) * strideElems + (lane >> 4) * 8;
    BFrag u;
    u.p.lo = *(const v8bf*)(p);
    u.p.hi = *(const v8bf*)(p + 16);
    return u.v;
}

// ---- Kernel 0: cast Wi / Ws to bf16 (they are already N-major: (4H, K)) ----
__global__ __launch_bounds__(256) void pack_weights(
    const float* __restrict__ Wi_f, const float* __restrict__ Ws_f,
    const float* __restrict__ Wi_b, const float* __restrict__ Ws_b,
    __bf16* __restrict__ WiP, __bf16* __restrict__ WsP)
{
    const int idx = blockIdx.x * 256 + threadIdx.x;
    if (idx < 2 * 512 * 128) {                       // WiP[dir][n=512][k=128]
        const int dir = idx >> 16;
        const float* W = dir ? Wi_b : Wi_f;
        WiP[idx] = (__bf16)W[idx & 65535];
    }
    if (idx < 2 * 512 * 256) {                       // WsP[dir][n=512][k=256]
        const int dir = idx >> 17;
        const float* W = dir ? Ws_b : Ws_f;
        WsP[idx] = (__bf16)W[idx & 131071];
    }
}

// ---- Kernel 1: Gx[dir][row][512] = LN(x @ Wi^T)*liw + lib  (fully parallel)
// rows = (b,i,j) flattened (65536), GEMM (65536x128)x(128x512), bf16 WMMA.
__global__ __launch_bounds__(256) void gx_kernel(
    const float* __restrict__ x, const __bf16* __restrict__ WiP,
    const float* __restrict__ liw_f, const float* __restrict__ lib_f,
    const float* __restrict__ liw_b, const float* __restrict__ lib_b,
    float* __restrict__ Gx)
{
    __shared__ __bf16 Ax[16][136];
    __shared__ union { __bf16 Bn[512][40]; float Cs[16][520]; } U;
    __shared__ float redS[16][16], redS2[16][16];

    const int tid  = threadIdx.x;
    const int dir  = blockIdx.y;
    const int r0   = blockIdx.x * 16;
    const int wave = tid >> 5, lane = tid & 31;
    const float* liw = dir ? liw_b : liw_f;
    const float* lib = dir ? lib_b : lib_f;
    const __bf16* Wg = WiP + (size_t)dir * 512 * 128;   // N-major (n,k)

    // stage A tile: 16 rows x 128, fp32 -> bf16
    {
        const int b = tid >> 4, cb = (tid & 15) * 8;
        const float* src = x + ((size_t)(r0 + b)) * 128 + cb;
        const float4 v0 = ((const float4*)src)[0];
        const float4 v1 = ((const float4*)src)[1];
        Ax[b][cb + 0] = (__bf16)v0.x; Ax[b][cb + 1] = (__bf16)v0.y;
        Ax[b][cb + 2] = (__bf16)v0.z; Ax[b][cb + 3] = (__bf16)v0.w;
        Ax[b][cb + 4] = (__bf16)v1.x; Ax[b][cb + 5] = (__bf16)v1.y;
        Ax[b][cb + 6] = (__bf16)v1.z; Ax[b][cb + 7] = (__bf16)v1.w;
    }

    const v8f vzero = {0.f, 0.f, 0.f, 0.f, 0.f, 0.f, 0.f, 0.f};
    v8f acc[4];
#pragma unroll
    for (int nt = 0; nt < 4; ++nt) acc[nt] = vzero;

    for (int ks = 0; ks < 4; ++ks) {            // K = 128 in chunks of 32
        const int kk = ks * 32;
        // stage B chunk N-major: 512 rows x 32 k-values (each thread: 2 rows)
#pragma unroll
        for (int rr = 0; rr < 2; ++rr) {
            const int n = tid * 2 + rr;
            const v8bf* src = (const v8bf*)&Wg[(size_t)n * 128 + kk];
            v8bf* dst = (v8bf*)&U.Bn[n][0];
            dst[0] = src[0]; dst[1] = src[1]; dst[2] = src[2]; dst[3] = src[3];
        }
        __syncthreads();
        const v16bf a = load_a_frag(&Ax[0][0], 136, kk, lane);
#pragma unroll
        for (int nt = 0; nt < 4; ++nt) {
            const int n0 = (wave * 4 + nt) * 16;
            const v16bf bf = load_b_frag_nmaj(&U.Bn[0][0], 40, n0, lane);
            acc[nt] = __builtin_amdgcn_wmma_f32_16x16x32_bf16(
                false, a, false, bf, (short)0, acc[nt], false, false);
        }
        __syncthreads();
    }

    // spill C (16x512 fp32) to LDS (C/D layout: row = r + 8*(lane>>4), col = n0+(lane&15))
    {
        const int n = lane & 15, rb = (lane >> 4) * 8;
#pragma unroll
        for (int nt = 0; nt < 4; ++nt) {
            const int nc = (wave * 4 + nt) * 16 + n;
#pragma unroll
            for (int r = 0; r < 8; ++r) U.Cs[rb + r][nc] = acc[nt][r];
        }
    }
    __syncthreads();

    // fused LayerNorm over 512 per row, then store fp32 Gx
    const int b = tid >> 4, seg = tid & 15;
    float s = 0.f, s2 = 0.f;
#pragma unroll 8
    for (int c0 = 0; c0 < 32; ++c0) { const float v = U.Cs[b][seg * 32 + c0]; s += v; s2 += v * v; }
    redS[b][seg] = s; redS2[b][seg] = s2;
    __syncthreads();
    float ts = 0.f, ts2 = 0.f;
#pragma unroll
    for (int k = 0; k < 16; ++k) { ts += redS[b][k]; ts2 += redS2[b][k]; }
    const float mean = ts * (1.f / 512.f);
    const float rstd = rsqrtf(ts2 * (1.f / 512.f) - mean * mean + 1e-5f);
    float* drow = Gx + ((size_t)dir * 65536 + (size_t)(r0 + b)) * 512;
#pragma unroll 8
    for (int c0 = 0; c0 < 32; ++c0) {
        const int c = seg * 32 + c0;
        drow[c] = (U.Cs[b][c] - mean) * rstd * liw[c] + lib[c];
    }
}

// ---- Kernel 2: one anti-diagonal wavefront step ----------------------------
// blockIdx.y = dir (0 fwd, 1 bwd). fwd processes diag i+j==t, bwd diag i+j==126-t.
// One block per cell: GEMM (16x256)x(256x512) with bf16 WMMA, fused LNs+gates.
__global__ __launch_bounds__(256) void step_kernel(
    float* __restrict__ out, const float* __restrict__ Gx,
    const __bf16* __restrict__ WsP, const unsigned char* __restrict__ masks,
    const float* __restrict__ lsw_f, const float* __restrict__ lsb_f,
    const float* __restrict__ lhw_f, const float* __restrict__ lhb_f,
    const float* __restrict__ lsw_b, const float* __restrict__ lsb_b,
    const float* __restrict__ lhw_b, const float* __restrict__ lhb_b,
    int t)
{
    __shared__ __bf16 Abf[16][264];   // A = [s0|s1] bf16, 16 x 256
    __shared__ union { __bf16 Bn[512][40]; float Cs[16][520]; } U;
    __shared__ float redS[16][16], redS2[16][16];

    const int tid = threadIdx.x;
    const int dir = blockIdx.y;
    const int d   = (dir == 0) ? t : (126 - t);
    const int i0  = (d > 63) ? (d - 63) : 0;
    const int ci  = i0 + (int)blockIdx.x;
    const int cj  = d - ci;
    const int wave = tid >> 5, lane = tid & 31;

    const float* lsw = dir ? lsw_b : lsw_f;
    const float* lsb = dir ? lsb_b : lsb_f;
    const float* lhw = dir ? lhw_b : lhw_f;
    const float* lhb = dir ? lhb_b : lhb_f;

    // neighbor cells (s0, s1) for this direction; boundaries read as zero
    int ri0, rj0, ri1, rj1; bool ok0, ok1;
    if (dir == 0) { ri0 = ci; rj0 = cj - 1; ok0 = (cj > 0);
                    ri1 = ci - 1; rj1 = cj; ok1 = (ci > 0); }
    else          { ri0 = ci; rj0 = cj + 1; ok0 = (cj < 63);
                    ri1 = ci + 1; rj1 = cj; ok1 = (ci < 63); }

    // stage A = [s0|s1] as bf16 from the state array (== output buffer)
    {
        const int b = tid >> 4, cb = (tid & 15) * 16;
        const bool isS0 = cb < 128;
        const int ri = isS0 ? ri0 : ri1, rj = isS0 ? rj0 : rj1;
        const bool valid = isS0 ? ok0 : ok1;
        const int cc = isS0 ? cb : (cb - 128);
        const float* src = out + (((size_t)b * NT + ri) * NT + rj) * N2H + (size_t)dir * NH + cc;
#pragma unroll
        for (int q = 0; q < 4; ++q) {
            const float4 v = valid ? ((const float4*)src)[q] : make_float4(0.f, 0.f, 0.f, 0.f);
            const int c = cb + q * 4;
            Abf[b][c]     = (__bf16)v.x; Abf[b][c + 1] = (__bf16)v.y;
            Abf[b][c + 2] = (__bf16)v.z; Abf[b][c + 3] = (__bf16)v.w;
        }
    }

    const v8f vzero = {0.f, 0.f, 0.f, 0.f, 0.f, 0.f, 0.f, 0.f};
    v8f acc[4];
#pragma unroll
    for (int nt = 0; nt < 4; ++nt) acc[nt] = vzero;

    const __bf16* Wg = WsP + (size_t)dir * 512 * 256;   // N-major (n,k)
    for (int ks = 0; ks < 8; ++ks) {            // K = 256 in chunks of 32
        const int kk = ks * 32;
        // stage B chunk N-major: 512 rows x 32 k-values (each thread: 2 rows)
#pragma unroll
        for (int rr = 0; rr < 2; ++rr) {
            const int n = tid * 2 + rr;
            const v8bf* src = (const v8bf*)&Wg[(size_t)n * 256 + kk];
            v8bf* dst = (v8bf*)&U.Bn[n][0];
            dst[0] = src[0]; dst[1] = src[1]; dst[2] = src[2]; dst[3] = src[3];
        }
        if (ks < 7)  // pull next Ws chunk toward the caches (global_prefetch_b8)
            __builtin_prefetch(&Wg[(size_t)(tid * 2) * 256 + kk + 32], 0, 1);
        __syncthreads();
        const v16bf a = load_a_frag(&Abf[0][0], 264, kk, lane);
#pragma unroll
        for (int nt = 0; nt < 4; ++nt) {
            const int n0 = (wave * 4 + nt) * 16;
            const v16bf bf = load_b_frag_nmaj(&U.Bn[0][0], 40, n0, lane);
            acc[nt] = __builtin_amdgcn_wmma_f32_16x16x32_bf16(
                false, a, false, bf, (short)0, acc[nt], false, false);
        }
        __syncthreads();
    }

    // spill raw Ws-matmul result (16x512 fp32) to LDS
    {
        const int n = lane & 15, rb = (lane >> 4) * 8;
#pragma unroll
        for (int nt = 0; nt < 4; ++nt) {
            const int nc = (wave * 4 + nt) * 16 + n;
#pragma unroll
            for (int r = 0; r < 8; ++r) U.Cs[rb + r][nc] = acc[nt][r];
        }
    }
    __syncthreads();

    // sg LayerNorm stats over 512 (on the raw matmul values)
    const int b = tid >> 4, seg = tid & 15;
    float s = 0.f, s2 = 0.f;
#pragma unroll 8
    for (int c0 = 0; c0 < 32; ++c0) { const float v = U.Cs[b][seg * 32 + c0]; s += v; s2 += v * v; }
    redS[b][seg] = s; redS2[b][seg] = s2;
    __syncthreads();
    float ts = 0.f, ts2 = 0.f;
#pragma unroll
    for (int k = 0; k < 16; ++k) { ts += redS[b][k]; ts2 += redS2[b][k]; }
    const float mean = ts * (1.f / 512.f);
    const float rstd = rsqrtf(ts2 * (1.f / 512.f) - mean * mean + 1e-5f);

    // re-read fp32 s0/s1 slices for this thread's 8 hidden indices (L2-hot)
    float s0v[8], s1v[8];
    {
        const float4 z4 = make_float4(0.f, 0.f, 0.f, 0.f);
        const float* p0 = out + (((size_t)b * NT + ri0) * NT + rj0) * N2H + (size_t)dir * NH + seg * 8;
        const float* p1 = out + (((size_t)b * NT + ri1) * NT + rj1) * N2H + (size_t)dir * NH + seg * 8;
        float4 a0 = ok0 ? ((const float4*)p0)[0] : z4;
        float4 a1 = ok0 ? ((const float4*)p0)[1] : z4;
        float4 b0 = ok1 ? ((const float4*)p1)[0] : z4;
        float4 b1 = ok1 ? ((const float4*)p1)[1] : z4;
        s0v[0]=a0.x; s0v[1]=a0.y; s0v[2]=a0.z; s0v[3]=a0.w;
        s0v[4]=a1.x; s0v[5]=a1.y; s0v[6]=a1.z; s0v[7]=a1.w;
        s1v[0]=b0.x; s1v[1]=b0.y; s1v[2]=b0.z; s1v[3]=b0.w;
        s1v[4]=b1.x; s1v[5]=b1.y; s1v[6]=b1.z; s1v[7]=b1.w;
    }

    // gates + h for 8 hidden indices per thread
    const size_t grow = ((size_t)dir * 65536 + ((size_t)b * 4096 + (size_t)ci * 64 + cj)) * 512;
    float hv[8];
    float hs = 0.f, hs2 = 0.f;
#pragma unroll
    for (int e = 0; e < 8; ++e) {
        const int hx = seg * 8 + e;
        const float sg0 = (U.Cs[b][hx]       - mean) * rstd * lsw[hx]       + lsb[hx];
        const float sg1 = (U.Cs[b][128 + hx] - mean) * rstd * lsw[128 + hx] + lsb[128 + hx];
        const float sg2 = (U.Cs[b][256 + hx] - mean) * rstd * lsw[256 + hx] + lsb[256 + hx];
        const float sg3 = (U.Cs[b][384 + hx] - mean) * rstd * lsw[384 + hx] + lsb[384 + hx];
        const float g0 = Gx[grow + hx]       + sg0;
        const float g1 = Gx[grow + 128 + hx] + sg1;
        const float g2 = Gx[grow + 256 + hx] + sg2;
        const float g3 = Gx[grow + 384 + hx] + sg3;
        const float r_inv = sigmoidf_(g0);
        const float ig    = sigmoidf_(g1);
        const float lg    = sigmoidf_(g3);
        const float nn    = tanhf(g2 - r_inv * sg2);
        const float h = nn + ig * (lg * s0v[e] + (1.f - lg) * s1v[e] - nn);
        hv[e] = h; hs += h; hs2 += h * h;
    }
    __syncthreads();                 // done reading redS from sg stats
    redS[b][seg] = hs; redS2[b][seg] = hs2;
    __syncthreads();
    float hts = 0.f, hts2 = 0.f;
#pragma unroll
    for (int k = 0; k < 16; ++k) { hts += redS[b][k]; hts2 += redS2[b][k]; }
    const float hmean = hts * (1.f / 128.f);
    const float hrstd = rsqrtf(hts2 * (1.f / 128.f) - hmean * hmean + 1e-5f);

    const float m = masks[(size_t)b * 4096 + (size_t)ci * 64 + cj] ? 1.f : 0.f;
    float* drow = out + (((size_t)b * NT + ci) * NT + cj) * N2H + (size_t)dir * NH;
#pragma unroll
    for (int e = 0; e < 8; ++e) {
        const int hx = seg * 8 + e;
        drow[hx] = ((hv[e] - hmean) * hrstd * lhw[hx] + lhb[hx]) * m;
    }
}

// ---- launch ----------------------------------------------------------------
extern "C" void kernel_launch(void* const* d_in, const int* in_sizes, int n_in,
                              void* d_out, int out_size, void* d_ws, size_t ws_size,
                              hipStream_t stream) {
    const float* x        = (const float*)d_in[0];
    const unsigned char* masks = (const unsigned char*)d_in[1];
    const float* Wi_f     = (const float*)d_in[2];
    const float* Ws_f     = (const float*)d_in[3];
    const float* lni_w_f  = (const float*)d_in[4];
    const float* lni_b_f  = (const float*)d_in[5];
    const float* lns_w_f  = (const float*)d_in[6];
    const float* lns_b_f  = (const float*)d_in[7];
    const float* lnh_w_f  = (const float*)d_in[8];
    const float* lnh_b_f  = (const float*)d_in[9];
    const float* Wi_b     = (const float*)d_in[10];
    const float* Ws_b     = (const float*)d_in[11];
    const float* lns_w_b  = (const float*)d_in[14];
    const float* lns_b_b  = (const float*)d_in[15];
    const float* lni_w_b  = (const float*)d_in[12];
    const float* lni_b_b  = (const float*)d_in[13];
    const float* lnh_w_b  = (const float*)d_in[16];
    const float* lnh_b_b  = (const float*)d_in[17];
    float* out = (float*)d_out;

    char* ws = (char*)d_ws;
    __bf16* WiP = (__bf16*)ws;                               // 2*512*128*2  = 256 KB
    __bf16* WsP = (__bf16*)(ws + 262144);                    // 2*512*256*2  = 512 KB
    float*  Gx  = (float*)(ws + 262144 + 524288);            // 2*65536*512*4 = 256 MB

    pack_weights<<<1024, 256, 0, stream>>>(Wi_f, Ws_f, Wi_b, Ws_b, WiP, WsP);
    gx_kernel<<<dim3(4096, 2), 256, 0, stream>>>(x, WiP,
        lni_w_f, lni_b_f, lni_w_b, lni_b_b, Gx);
    for (int t = 0; t < 127; ++t) {
        const int L = (t <= 63) ? (t + 1) : (127 - t);
        step_kernel<<<dim3(L, 2), 256, 0, stream>>>(out, Gx, WsP, masks,
            lns_w_f, lns_b_f, lnh_w_f, lnh_b_f,
            lns_w_b, lns_b_b, lnh_w_b, lnh_b_b, t);
    }
}
